// JukeboxAutoEncoder_41446434406495
// MI455X (gfx1250) — matrix operand order repaired and checked by reading, not compiled, not run on hardware
//
#include <hip/hip_runtime.h>
#include <hip/hip_bf16.h>

// ---------------------------------------------------------------------------
// JukeboxAutoEncoder on MI455X (gfx1250, wave32, WMMA)
//
// - All Cin=Cout=64 convs + codebook GEMM on v_wmma_f32_16x16x32_f16.
// - Activations kept f16 end-to-end (roofline: halves conv-stack HBM traffic;
//   f16 WMMA compute is far below the 23.3 TB/s memory bound).
// - Weights/codebook pre-packed so B fragments are contiguous B128 loads.
// - Conv kernel templated on <KEFF,STRIDE,OSTRIDE,RES> -> branch-free epilogue.
// - Quantise fuses similarity (non-temporal 537MB f32 stream) + wave32 argmin.
// ---------------------------------------------------------------------------

typedef __attribute__((ext_vector_type(16))) _Float16 v16h;
typedef __attribute__((ext_vector_type(8)))  float    v8f;

#define WMMA_F16(a, b, c) \
  __builtin_amdgcn_wmma_f32_16x16x32_f16(false, (a), false, (b), (short)0, (c), false, false)

__device__ __forceinline__ unsigned pack_h2(float a, float b) {
  union { _Float16 h[2]; unsigned u; } p;
  p.h[0] = (_Float16)a; p.h[1] = (_Float16)b;
  return p.u;
}

// K offset inside a 32-wide chunk for A fragment register r (contiguous: k = 16*half + 2r).
__device__ __forceinline__ int a_koff(int r, int half) {
  int base = r << 1;
  return base + (half ? 16 : 0);
}

// ---------------------------------------------------------------------------
// Weight pre-pack: w f32 [K,64,64] -> u32 [64][Keff*32]: wpk[n*P + p] holds the
// f16 pair (B[2p][n], B[2p+1][n]), B[i*64+ci][n] = w[k_start+i*k_step][ci][n].
// Layout is co-major so each B fragment chunk is 8 contiguous dwords (2xB128).
// ---------------------------------------------------------------------------
__global__ void pack_w16(const float* __restrict__ w, unsigned* __restrict__ wpk,
                         int k_start, int k_step, int Keff) {
  int idx = blockIdx.x * 256 + threadIdx.x;
  int P = Keff << 5;
  if (idx >= P * 64) return;
  int n = idx / P, p = idx - n * P;
  int kk = p << 1;
  int i_tap = kk >> 6, ci = kk & 63;
  long base = ((long)(k_start + i_tap * k_step) * 64 + ci) * 64 + n;
  wpk[idx] = pack_h2(w[base], w[base + 64]);
}

// ---------------------------------------------------------------------------
// Generic Cin=Cout=64 conv as WMMA GEMM (f16 in / f16 out, f32 accumulate).
// out[b, t*OS+out_off, co] = relu( bias[co]
//   + sum_{i<KEFF,ci} x[b, t*STRIDE - padL + i*dil, ci] * W[i,ci,co] ) (+resid)
// Block: 256 thr (8 waves) = 128 output positions x 64 cout.
// ---------------------------------------------------------------------------
template <int KEFF, int STRIDE, int OS, bool RES>
__global__ __launch_bounds__(256) void conv64_wmma(
    const _Float16* __restrict__ x, const unsigned* __restrict__ wpk,
    const float* __restrict__ bias, const _Float16* __restrict__ resid,
    _Float16* __restrict__ out, int Tin, int ToutIdx, int outT,
    int dil, int padL, int out_off) {
  extern __shared__ char smem[];
  _Float16* sX = (_Float16*)smem;                     // [span][64] f16
  const int tilesPerB = ToutIdx >> 7;
  const int b  = blockIdx.x / tilesPerB;
  const int t0 = (blockIdx.x - b * tilesPerB) << 7;
  const int tid = threadIdx.x;

  const int span = 127 * STRIDE + (KEFF - 1) * dil + 1;
  const int tinBase = t0 * STRIDE - padL;
  // Stage input rows (full-row B128 copies; rows out of range -> zeros).
  uint4* sX4 = (uint4*)sX;
  const uint4* x4 = (const uint4*)x;
  const long xrow0 = ((long)b * Tin + tinBase) << 3;  // uint4 units (8 halfs)
  for (int i = tid; i < span * 8; i += 256) {
    int r = i >> 3;
    int t = tinBase + r;
    uint4 v = {0u, 0u, 0u, 0u};
    if (t >= 0 && t < Tin) v = x4[xrow0 + i];
    sX4[i] = v;
  }
  __syncthreads();

  const int wave = tid >> 5, lane = tid & 31;
  const int half = lane >> 4, ln = lane & 15;
  const int mRow = (wave << 4) + ln;                  // A-matrix row (time)
  const uint4* w4 = (const uint4*)wpk;

  for (int ntile = 0; ntile < 4; ++ntile) {
    const int co = (ntile << 4) + ln;
    const int wbase4 = co * (KEFF << 3) + (half << 1);  // uint4 units
    v8f acc = {0.f, 0.f, 0.f, 0.f, 0.f, 0.f, 0.f, 0.f};
#pragma unroll
    for (int kc = 0; kc < 2 * KEFF; ++kc) {
      union { v16h v; unsigned u[8]; } af;
#pragma unroll
      for (int r = 0; r < 8; ++r) {
        int kk = (kc << 5) + a_koff(r, half);
        int i_tap = kk >> 6, ci = kk & 63;
        int rr = mRow * STRIDE + i_tap * dil;
        af.u[r] = *(const unsigned*)&sX[rr * 64 + ci];
      }
      union { v16h v; uint4 q[2]; } bf;
      bf.q[0] = w4[wbase4 + (kc << 2)];
      bf.q[1] = w4[wbase4 + (kc << 2) + 1];
      acc = WMMA_F16(af.v, bf.v, acc);
    }
    const float bs = bias[co];
    const int m0 = (wave << 4) + (half << 3);
    const long obase = ((long)b * outT + (long)(t0 + m0) * OS + out_off) * 64 + co;
#pragma unroll
    for (int r = 0; r < 8; ++r) {
      long oidx = obase + (long)r * OS * 64;
      float v = fmaxf(acc[r] + bs, 0.f);
      if (RES) v += (float)resid[oidx];
      out[oidx] = (_Float16)v;
    }
  }
}

// ---------------------------------------------------------------------------
// First encoder conv: Cin=1, K=4, stride=2, SAME (padL=1), ReLU -> f16 out.
// ---------------------------------------------------------------------------
__global__ void conv_down_c1(const float* __restrict__ x, const float* __restrict__ wd,
                             const float* __restrict__ bd, _Float16* __restrict__ out,
                             int Tin, int Tout) {
  long o = (long)blockIdx.x * 256 + threadIdx.x;
  int co = (int)(o & 63);
  long bt = o >> 6;
  int t = (int)(bt % Tout);
  int b = (int)(bt / Tout);
  float acc = bd[co];
#pragma unroll
  for (int k = 0; k < 4; ++k) {
    int tin = 2 * t - 1 + k;
    if (tin >= 0 && tin < Tin) acc += x[(long)b * Tin + tin] * wd[k * 64 + co];
  }
  out[o] = (_Float16)fmaxf(acc, 0.f);
}

// ---------------------------------------------------------------------------
// Final conv: 64 -> 1, K=3, SAME, no activation. f16 in, f32 out (x_hat).
// ---------------------------------------------------------------------------
__global__ void conv_out_c1(const _Float16* __restrict__ x, const float* __restrict__ w,
                            const float* __restrict__ bb, float* __restrict__ out, int T) {
  long o = (long)blockIdx.x * 256 + threadIdx.x;      // over B*T
  int t = (int)(o % T);
  long b = o / T;
  float acc = bb[0];
  for (int k = 0; k < 3; ++k) {
    int tin = t - 1 + k;
    if (tin < 0 || tin >= T) continue;
    const _Float16* xr = x + (((long)b * T + tin) << 6);
#pragma unroll
    for (int ci = 0; ci < 64; ++ci) acc += (float)xr[ci] * w[k * 64 + ci];
  }
  out[o] = acc;
}

// ---------------------------------------------------------------------------
// Codebook prep: f32 [2048,64] -> f16 copy + squared norms + rsqrt norms.
// ---------------------------------------------------------------------------
__global__ void cb_prep(const float* __restrict__ cb, _Float16* __restrict__ cbh,
                        float* __restrict__ nsq, float* __restrict__ rsq) {
  int e = blockIdx.x * 256 + threadIdx.x;
  if (e >= 2048) return;
  float acc = 0.f;
  for (int c = 0; c < 64; ++c) {
    float v = cb[e * 64 + c];
    cbh[e * 64 + c] = (_Float16)v;
    acc += v * v;
  }
  nsq[e] = acc;
  rsq[e] = rsqrtf(fmaxf(acc, 1e-30f));
}

// ---------------------------------------------------------------------------
// Quantise: WMMA GEMM z[65536,64] x cb^T[64,2048]; fused similarity stream
// (non-temporal) + argmin (shfl_xor over the 16-lane N group). 128 rows/block.
// ---------------------------------------------------------------------------
__global__ __launch_bounds__(256) void quantize_wmma(
    const _Float16* __restrict__ z, const unsigned* __restrict__ cbh,
    const float* __restrict__ nsqE, const float* __restrict__ rsqE,
    float* __restrict__ sim, int* __restrict__ ids) {
  __shared__ _Float16 sZ[128 * 64];
  __shared__ float sNsq[128];
  __shared__ float sRsq[128];
  const int tid = threadIdx.x;
  const long r0 = (long)blockIdx.x << 7;

  for (int i = tid; i < 128 * 8; i += 256)
    ((uint4*)sZ)[i] = ((const uint4*)z)[(r0 << 3) + i];
  __syncthreads();
  if (tid < 128) {
    float acc = 0.f;
    for (int c = 0; c < 64; ++c) { float v = (float)sZ[tid * 64 + c]; acc += v * v; }
    sNsq[tid] = acc;
    sRsq[tid] = rsqrtf(fmaxf(acc, 1e-30f));
  }
  __syncthreads();

  const int wave = tid >> 5, lane = tid & 31;
  const int half = lane >> 4, ln = lane & 15;
  const int mRow = (wave << 4) + ln;
  const uint4* cb4 = (const uint4*)cbh;

  union { v16h v; unsigned u[8]; } a0, a1;
#pragma unroll
  for (int r = 0; r < 8; ++r) {
    int kk = a_koff(r, half);
    a0.u[r] = *(const unsigned*)&sZ[mRow * 64 + kk];
    a1.u[r] = *(const unsigned*)&sZ[mRow * 64 + 32 + kk];
  }
  float nz[8], rz[8], best[8];
  int bid[8];
#pragma unroll
  for (int r = 0; r < 8; ++r) {
    int m = (wave << 4) + r + (half << 3);
    nz[r] = sNsq[m]; rz[r] = sRsq[m];
    best[r] = 3.4e38f; bid[r] = 0;
  }

  for (int et = 0; et < 128; ++et) {
    int eL = (et << 4) + ln;
    if (et < 127) __builtin_prefetch(cbh + ((size_t)(eL + 16) << 5), 0, 0);
    union { v16h v; uint4 q[2]; } b0, b1;
    const int ebase = (eL << 3) + (half << 1);   // uint4 units into [e][32 dwords]
    b0.q[0] = cb4[ebase];     b0.q[1] = cb4[ebase + 1];
    b1.q[0] = cb4[ebase + 4]; b1.q[1] = cb4[ebase + 5];
    v8f c = {0.f, 0.f, 0.f, 0.f, 0.f, 0.f, 0.f, 0.f};
    c = WMMA_F16(a0.v, b0.v, c);
    c = WMMA_F16(a1.v, b1.v, c);
    float ne = nsqE[eL], re = rsqE[eL];
#pragma unroll
    for (int r = 0; r < 8; ++r) {
      float dot = c[r];
      long row = r0 + (wave << 4) + r + (half << 3);
      __builtin_nontemporal_store(dot * rz[r] * re, &sim[row * 2048 + eL]);
      float dist = ne + nz[r] - 2.f * dot;
      if (dist < best[r]) { best[r] = dist; bid[r] = eL; }
    }
  }
#pragma unroll
  for (int r = 0; r < 8; ++r) {
    float v = best[r]; int id = bid[r];
    for (int mask = 8; mask >= 1; mask >>= 1) {
      float ov = __shfl_xor(v, mask, 32);
      int oid = __shfl_xor(id, mask, 32);
      if (ov < v || (ov == v && oid < id)) { v = ov; id = oid; }
    }
    if (ln == 0) {
      long row = r0 + (wave << 4) + r + (half << 3);
      ids[row] = id;
    }
  }
}

// ---------------------------------------------------------------------------
// z_q gather: f32 z_q to d_out, f16 copy for the decoder, per-row ||z - z_q||.
// ---------------------------------------------------------------------------
__global__ __launch_bounds__(256) void zq_gather(
    const int* __restrict__ ids, const float* __restrict__ cb,
    const _Float16* __restrict__ z, float* __restrict__ zq,
    _Float16* __restrict__ zqh, float* __restrict__ rowNorm) {
  __shared__ float sAcc[256];
  int tid = threadIdx.x;
  long row = (long)blockIdx.x * 4 + (tid >> 6);
  int c = tid & 63;
  int id = ids[row];
  float q = cb[id * 64 + c];
  zq[row * 64 + c] = q;
  zqh[row * 64 + c] = (_Float16)q;
  float d = (float)z[row * 64 + c] - q;
  sAcc[tid] = d * d;
  __syncthreads();
  if (c == 0) {
    float s = 0.f;
    for (int k = 0; k < 64; ++k) s += sAcc[(tid & 192) + k];
    rowNorm[row] = sqrtf(s);
  }
}

// Deterministic fixed-order loss reduction: loss = (1+BETA) * mean(rowNorm)
__global__ __launch_bounds__(256) void loss_reduce(const float* __restrict__ rowNorm,
                                                   float* __restrict__ lossOut, int n) {
  __shared__ float sAcc[256];
  int tid = threadIdx.x;
  float s = 0.f;
  for (int i = tid; i < n; i += 256) s += rowNorm[i];
  sAcc[tid] = s;
  __syncthreads();
  for (int w = 128; w > 0; w >>= 1) {
    if (tid < w) sAcc[tid] += sAcc[tid + w];
    __syncthreads();
  }
  if (tid == 0) *lossOut = sAcc[0] * (1.25f / (float)n);
}

// ---------------------------------------------------------------------------
// Host-side conv dispatcher (selects template instantiation)
// ---------------------------------------------------------------------------
static void launch_conv(int Keff, int stride, int os, bool res,
                        const _Float16* x, const unsigned* wpk, const float* bias,
                        const _Float16* resid, _Float16* out,
                        int Tin, int ToutIdx, int outT, int dil, int padL, int outOff,
                        hipStream_t stream) {
  int span = 127 * stride + (Keff - 1) * dil + 1;
  size_t sh = (size_t)span * 64 * sizeof(_Float16);
  int blocks = 4 * (ToutIdx >> 7);
  if (Keff == 3 && !res)
    conv64_wmma<3, 1, 1, false><<<blocks, 256, sh, stream>>>(x, wpk, bias, resid, out,
        Tin, ToutIdx, outT, dil, padL, outOff);
  else if (Keff == 3 && res)
    conv64_wmma<3, 1, 1, true><<<blocks, 256, sh, stream>>>(x, wpk, bias, resid, out,
        Tin, ToutIdx, outT, dil, padL, outOff);
  else if (Keff == 4)
    conv64_wmma<4, 2, 1, false><<<blocks, 256, sh, stream>>>(x, wpk, bias, resid, out,
        Tin, ToutIdx, outT, dil, padL, outOff);
  else  // Keff == 2: transposed-conv phase
    conv64_wmma<2, 1, 2, false><<<blocks, 256, sh, stream>>>(x, wpk, bias, resid, out,
        Tin, ToutIdx, outT, dil, padL, outOff);
}

// ---------------------------------------------------------------------------
// Host orchestration
// ---------------------------------------------------------------------------
extern "C" void kernel_launch(void* const* d_in, const int* in_sizes, int n_in,
                              void* d_out, int out_size, void* d_ws, size_t ws_size,
                              hipStream_t stream) {
  (void)in_sizes; (void)n_in; (void)out_size; (void)ws_size;
  const int B = 4, T0 = 131072;
  auto F = [&](int i) { return (const float*)d_in[i]; };
  // Leaf order: x | 3x enc{wd,bd,2x(w1,b1,w2,b2)} | codebook |
  //             3x dec{2x(w1,b1,w2,b2),wu,bu} | w_out | b_out
  const int IX_CB = 31, IX_WOUT = 62, IX_BOUT = 63;
  auto encB = [&](int L) { return 1 + L * 10; };
  auto decB = [&](int L) { return 32 + L * 10; };

  // d_out layout (float units): x_hat | loss | z_q | similarity | ids
  float* o = (float*)d_out;
  const long XHAT = 0;
  const long LOSS = XHAT + (long)B * T0;
  const long ZQ   = LOSS + 1;
  const long SIM  = ZQ + (long)B * 16384 * 64;
  const long IDS  = SIM + (long)B * 16384 * 2048;

  // workspace layout (f16 activation buffers)
  char* ws = (char*)d_ws;
  const size_t BIG = (size_t)B * T0 * 64 * sizeof(_Float16);  // 64 MB
  _Float16* bufA = (_Float16*)ws;
  _Float16* bufB = (_Float16*)(ws + BIG);
  char* small = ws + 2 * BIG;
  _Float16* cbh = (_Float16*)small;  small += 2048 * 64 * sizeof(_Float16);
  float* nsqE = (float*)small;       small += 2048 * sizeof(float);
  float* rsqE = (float*)small;       small += 2048 * sizeof(float);
  float* rowNorm = (float*)small;    small += 65536 * sizeof(float);
  _Float16* zqh = (_Float16*)small;  small += (size_t)65536 * 64 * sizeof(_Float16);
  unsigned* wcur = (unsigned*)small;

  auto pack = [&](const float* w, int ks, int kstep, int Keff) -> unsigned* {
    unsigned* dst = wcur;
    int total = (Keff << 5) * 64;
    wcur += total;
    pack_w16<<<(total + 255) / 256, 256, 0, stream>>>(w, dst, ks, kstep, Keff);
    return dst;
  };

  cb_prep<<<8, 256, 0, stream>>>(F(IX_CB), cbh, nsqE, rsqE);

  // ---------------- Encoder ----------------
  {
    long n = (long)B * 65536 * 64;
    conv_down_c1<<<(int)(n / 256), 256, 0, stream>>>(F(0), F(encB(0)), F(encB(0) + 1),
                                                     bufA, T0, 65536);
  }
  _Float16* cur = bufA;
  _Float16* tmp = bufB;
  int T = 65536;
  for (int L = 0; L < 3; ++L) {
    int eb = encB(L);
    if (L > 0) {  // strided down conv, Cin=64
      unsigned* wd = pack(F(eb), 0, 1, 4);
      launch_conv(4, 2, 1, false, cur, wd, F(eb + 1), nullptr, tmp,
                  T, T / 2, T / 2, 1, 1, 0, stream);
      _Float16* s = cur; cur = tmp; tmp = s;
      T /= 2;
    }
    for (int r = 0; r < 2; ++r) {
      const float* w1 = F(eb + 2 + 4 * r), *b1 = F(eb + 3 + 4 * r);
      const float* w2 = F(eb + 4 + 4 * r), *b2 = F(eb + 5 + 4 * r);
      unsigned* w1p = pack(w1, 0, 1, 3);
      unsigned* w2p = pack(w2, 0, 1, 3);
      launch_conv(3, 1, 1, false, cur, w1p, b1, nullptr, tmp, T, T, T, 3, 3, 0, stream);
      launch_conv(3, 1, 1, true,  tmp, w2p, b2, cur, cur,   T, T, T, 1, 1, 0, stream);
    }
  }
  // cur = z (f16): [B,16384,64] = 65536 rows

  // ---------------- Quantise ----------------
  int* ids = (int*)(o + IDS);
  quantize_wmma<<<512, 256, 0, stream>>>(cur, (const unsigned*)cbh, nsqE, rsqE,
                                         o + SIM, ids);
  zq_gather<<<16384, 256, 0, stream>>>(ids, F(IX_CB), cur, o + ZQ, zqh, rowNorm);
  loss_reduce<<<1, 256, 0, stream>>>(rowNorm, o + LOSS, 65536);

  // ---------------- Decoder ----------------
  const _Float16* zcur = zqh;
  _Float16* zbuf = bufA;
  _Float16* hbuf = bufB;
  T = 16384;
  for (int L = 0; L < 3; ++L) {
    int db = decB(L);
    for (int r = 0; r < 2; ++r) {
      const float* w1 = F(db + 4 * r), *b1 = F(db + 1 + 4 * r);
      const float* w2 = F(db + 2 + 4 * r), *b2 = F(db + 3 + 4 * r);
      unsigned* w1p = pack(w1, 0, 1, 3);
      unsigned* w2p = pack(w2, 0, 1, 3);
      launch_conv(3, 1, 1, false, zcur, w1p, b1, nullptr, hbuf, T, T, T, 1, 1, 0, stream);
      launch_conv(3, 1, 1, true,  hbuf, w2p, b2, zcur, zbuf,  T, T, T, 3, 3, 0, stream);
      zcur = zbuf;
    }
    // Transposed conv (stride 2, K=4, SAME):
    //   out[2t] = x[t-1]w0 + x[t]w2 ; out[2t+1] = x[t]w1 + x[t+1]w3
    unsigned* we = pack(F(db + 8), 0, 2, 2);
    unsigned* wo = pack(F(db + 8), 1, 2, 2);
    launch_conv(2, 1, 2, false, zcur, we, F(db + 9), nullptr, hbuf,
                T, T, 2 * T, 1, 1, 0, stream);
    launch_conv(2, 1, 2, false, zcur, wo, F(db + 9), nullptr, hbuf,
                T, T, 2 * T, 1, 0, 1, stream);
    zcur = hbuf;
    _Float16* s = zbuf; zbuf = hbuf; hbuf = s;
    T *= 2;
  }

  // ---------------- Output conv (64 -> 1) ----------------
  {
    long n = (long)B * T0;
    conv_out_c1<<<(int)(n / 256), 256, 0, stream>>>(zcur, F(IX_WOUT), F(IX_BOUT),
                                                    o + XHAT, T0);
  }
}
